// SchNet_67946382623316
// MI455X (gfx1250) — compile-verified, matrix operand order checked by reading, not run on hardware
//
#include <hip/hip_runtime.h>
#include <hip/hip_bf16.h>
#include <math.h>

typedef __attribute__((ext_vector_type(16))) _Float16 v16h;
typedef __attribute__((ext_vector_type(8)))  float    v8f;

#define NATOMS 30000
#define NEDGES 600000
#define DD 128
#define GG 32
#define LN2F 0.69314718055994530942f

// ssp(x) = softplus(x) - ln 2, stable max-form with fast exp/log
// (v_exp_f32 + v_log_f32 TRANS ops instead of the precise log1pf expansion)
__device__ __forceinline__ float sspf(float x) {
    float ax = fabsf(x);
    return fmaxf(x, 0.0f) + __logf(1.0f + __expf(-ax)) - LN2F;
}

// ---------------------------------------------------------------------------
// WMMA fragment helpers (CDNA5 16x16x32 f16 layout, wave32):
//   A (16x32, f16): lane l -> row (l&15); f16 elem e:
//       e<8  : K = hi*8 + e          (hi = l>>4)
//       e>=8 : K = 16 + hi*8 + (e-8)
//   B (32x16, f16): lane l -> col (l&15), same K packing (column-major read)
//   C (16x16, f32): vgpr i: lanes 0-15 -> M=i, lanes 16-31 -> M=8+i; N=l&15
// ---------------------------------------------------------------------------

// A fragment from a row-major f32 matrix (row stride ld), K-chunk at kbase.
__device__ __forceinline__ v16h a_frag_f32(const float* __restrict__ X, int ld,
                                           int row, int kbase, int hi) {
    const float* p0 = X + (size_t)row * ld + kbase + hi * 8;
    const float* p1 = p0 + 16;
    v16h a;
#pragma unroll
    for (int e = 0; e < 8; ++e) {
        a[e]     = (_Float16)p0[e];
        a[e + 8] = (_Float16)p1[e];
    }
    return a;
}

// B fragment straight from a row-major f32 KxN weight (K=32 single k-step).
__device__ __forceinline__ v16h b_frag_f32(const float* __restrict__ W, int N,
                                           int t, int lane) {
    int hi = lane >> 4;
    int n  = t * 16 + (lane & 15);
    v16h b;
#pragma unroll
    for (int e = 0; e < 8; ++e) {
        b[e]     = (_Float16)W[(hi * 8 + e) * N + n];
        b[e + 8] = (_Float16)W[(16 + hi * 8 + e) * N + n];
    }
    return b;
}

static __device__ __forceinline__ v8f vzero8() {
    v8f z = {0.f, 0.f, 0.f, 0.f, 0.f, 0.f, 0.f, 0.f};
    return z;
}

// ---------------------------------------------------------------------------
// Generic fused GEMM:  Y = act(X @ W + bias) [+ res]
// X: M x K f32 (M = rowtiles*16), W: K x N f32 row-major.
// NT = N/16 tiles, KS = K/32 k-steps. 4 waves per block, one 16-row tile
// per wave per grid-stride iteration. W is pre-swizzled to f16 fragment
// order in LDS so each wmma consumes one contiguous 32B LDS read.
// ---------------------------------------------------------------------------
template <int NT, int KS, int ACT, bool RES>
__global__ __launch_bounds__(128) void wmma_gemm_kernel(
    const float* __restrict__ X, const float* __restrict__ W,
    const float* __restrict__ bias, const float* __restrict__ res,
    float* __restrict__ Y, int rowtiles) {
    constexpr int N = NT * 16;
    constexpr int K = KS * 32;
    __shared__ alignas(32) _Float16 sW[NT * KS * 512];

    int tid = threadIdx.x;
    // Cooperative f32 -> f16 swizzle of W into fragment order.
    for (int idx = tid; idx < NT * KS * 512; idx += blockDim.x) {
        int e    = idx & 15;
        int lane = (idx >> 4) & 31;
        int grp  = idx >> 9;
        int ks   = grp % KS;
        int t    = grp / KS;
        int hi   = lane >> 4;
        int n    = t * 16 + (lane & 15);
        int k    = ks * 32 + ((e < 8) ? (hi * 8 + e) : (16 + hi * 8 + (e - 8)));
        sW[idx]  = (_Float16)W[k * N + n];
    }
    __syncthreads();

    int lane = tid & 31;
    int wave = tid >> 5;
    int hi   = lane >> 4;
    int lo   = lane & 15;

    for (int rt = blockIdx.x * 4 + wave; rt < rowtiles; rt += gridDim.x * 4) {
        int rowbase = rt * 16;
        // Prefetch next tile's A rows into cache (global_prefetch_b8).
        int rtn = rt + gridDim.x * 4;
        if (rtn < rowtiles)
            __builtin_prefetch(X + (size_t)rtn * 16 * K + lane * (16 * K / 32),
                               0, 1);

        v8f c[NT];
#pragma unroll
        for (int t = 0; t < NT; ++t) c[t] = vzero8();

#pragma unroll
        for (int ks = 0; ks < KS; ++ks) {
            v16h a = a_frag_f32(X, K, rowbase + lo, ks * 32, hi);
#pragma unroll
            for (int t = 0; t < NT; ++t) {
                v16h b = *(const v16h*)&sW[((t * KS + ks) * 32 + lane) * 16];
                c[t] = __builtin_amdgcn_wmma_f32_16x16x32_f16(
                    false, a, false, b, (short)0, c[t], false, false);
            }
        }
#pragma unroll
        for (int t = 0; t < NT; ++t) {
            int   col = t * 16 + lo;
            float bb  = bias[col];
#pragma unroll
            for (int i = 0; i < 8; ++i) {
                int   row = rowbase + i + hi * 8;
                float v   = c[t][i] + bb;
                if (ACT) v = sspf(v);
                if (RES) v += res[(size_t)row * N + col];
                Y[(size_t)row * N + col] = v;
            }
        }
    }
}

// ---------------------------------------------------------------------------
// Fused edge kernel. Each wave owns a 16-edge tile:
//   smear (built analytically in A-fragment layout) -> WMMA x2 (K=32,N=32)
//   -> ssp via per-wave LDS bounce -> WMMA x8 (K=32,N=128) + be2
//   -> gather rn[a0], rn[a1] and scatter both message directions with
//      global_atomic_add_f32 into m.
// ---------------------------------------------------------------------------
__global__ __launch_bounds__(128) void edge_kernel(
    const float* __restrict__ xyz, const int* __restrict__ nbr,
    const float* __restrict__ We1, const float* __restrict__ be1,
    const float* __restrict__ We2, const float* __restrict__ be2,
    const float* __restrict__ rn, float* __restrict__ mbuf, int ntiles) {
    __shared__ float hld[4][16][GG];

    int tid  = threadIdx.x;
    int lane = tid & 31;
    int wave = tid >> 5;
    int hi   = lane >> 4;
    int lo   = lane & 15;

    const float width = 5.0f / 31.0f;            // CUTOFF/(G-1)
    const float coeff = -0.5f / (width * width);

    // Hoist all B fragments + biases for the whole loop.
    v16h bW1_0 = b_frag_f32(We1, GG, 0, lane);
    v16h bW1_1 = b_frag_f32(We1, GG, 1, lane);
    v16h bW2[8];
#pragma unroll
    for (int t = 0; t < 8; ++t) bW2[t] = b_frag_f32(We2, DD, t, lane);
    float be1a = be1[lo], be1b = be1[16 + lo];
    float be2v[8];
#pragma unroll
    for (int t = 0; t < 8; ++t) be2v[t] = be2[t * 16 + lo];

    for (int et = blockIdx.x * 4 + wave; et < ntiles; et += gridDim.x * 4) {
        int e0 = et * 16;
        // Prefetch next tile's neighbor list.
        int etn = et + gridDim.x * 4;
        if (etn < ntiles) __builtin_prefetch(nbr + 2 * etn * 16, 0, 1);

        // Each lane computes the distance of edge (e0+lo).
        int   ia = nbr[2 * (e0 + lo)], ib = nbr[2 * (e0 + lo) + 1];
        float dx = xyz[3 * ia + 0] - xyz[3 * ib + 0];
        float dy = xyz[3 * ia + 1] - xyz[3 * ib + 1];
        float dz = xyz[3 * ia + 2] - xyz[3 * ib + 2];
        float d  = sqrtf(dx * dx + dy * dy + dz * dz);

        // Gaussian smear directly in A-fragment layout (no memory round-trip).
        v16h a;
#pragma unroll
        for (int el = 0; el < 8; ++el) {
            float t0 = d - (float)(hi * 8 + el) * width;
            a[el]    = (_Float16)__expf(coeff * t0 * t0);
            float t1 = d - (float)(16 + hi * 8 + el) * width;
            a[el + 8] = (_Float16)__expf(coeff * t1 * t1);
        }

        v8f c0 = vzero8(), c1 = vzero8();
        c0 = __builtin_amdgcn_wmma_f32_16x16x32_f16(false, a, false, bW1_0,
                                                    (short)0, c0, false, false);
        c1 = __builtin_amdgcn_wmma_f32_16x16x32_f16(false, a, false, bW1_1,
                                                    (short)0, c1, false, false);

        // ssp + be1 -> LDS (C layout) -> reload as A fragment.
#pragma unroll
        for (int i = 0; i < 8; ++i) {
            int row = i + hi * 8;
            hld[wave][row][lo]      = sspf(c0[i] + be1a);
            hld[wave][row][16 + lo] = sspf(c1[i] + be1b);
        }
        asm volatile("s_wait_dscnt 0" ::: "memory");
        v16h a2;
#pragma unroll
        for (int el = 0; el < 8; ++el) {
            a2[el]     = (_Float16)hld[wave][lo][hi * 8 + el];
            a2[el + 8] = (_Float16)hld[wave][lo][16 + hi * 8 + el];
        }

        v8f w[8];
#pragma unroll
        for (int t = 0; t < 8; ++t) {
            w[t] = vzero8();
            w[t] = __builtin_amdgcn_wmma_f32_16x16x32_f16(
                false, a2, false, bW2[t], (short)0, w[t], false, false);
        }

        // Scatter. C layout: vgpr i of lane -> edge (e0 + i + hi*8), feat t*16+lo.
        int ea[8], eb[8];
#pragma unroll
        for (int i = 0; i < 8; ++i) {
            int ee = e0 + i + hi * 8;
            ea[i]  = nbr[2 * ee];
            eb[i]  = nbr[2 * ee + 1];
        }
#pragma unroll
        for (int t = 0; t < 8; ++t) {
            int f = t * 16 + lo;
#pragma unroll
            for (int i = 0; i < 8; ++i) {
                float wv = w[t][i] + be2v[t];
                float r0 = rn[(size_t)ea[i] * DD + f];
                float r1 = rn[(size_t)eb[i] * DD + f];
                atomicAdd(&mbuf[(size_t)eb[i] * DD + f], r0 * wv);
                atomicAdd(&mbuf[(size_t)ea[i] * DD + f], r1 * wv);
            }
        }
    }
}

// r[a][f] = embed[z[a]][f]
__global__ void embed_kernel(const int* __restrict__ z,
                             const float* __restrict__ embed,
                             float* __restrict__ r, int n) {
    int idx = blockIdx.x * blockDim.x + threadIdx.x;
    if (idx < n) {
        int a = idx >> 7, f = idx & 127;
        r[idx] = embed[(size_t)z[a] * DD + f];
    }
}

// One wave per atom: e = h[atom] . Wr2 + br2 ; energy[mol_idx[atom]] += e
__global__ void readout_kernel(const float* __restrict__ h,
                               const float* __restrict__ Wr2,
                               const float* __restrict__ br2,
                               const int* __restrict__ mol_idx,
                               float* __restrict__ energy, int natoms) {
    int gw   = (blockIdx.x * blockDim.x + threadIdx.x) >> 5;
    int lane = threadIdx.x & 31;
    if (gw >= natoms) return;
    float s = h[(size_t)gw * 64 + lane] * Wr2[lane] +
              h[(size_t)gw * 64 + 32 + lane] * Wr2[32 + lane];
#pragma unroll
    for (int off = 16; off > 0; off >>= 1) s += __shfl_down(s, off, 32);
    if (lane == 0) atomicAdd(&energy[mol_idx[gw]], s + br2[0]);
}

extern "C" void kernel_launch(void* const* d_in, const int* in_sizes, int n_in,
                              void* d_out, int out_size, void* d_ws,
                              size_t ws_size, hipStream_t stream) {
    const int*   z       = (const int*)d_in[0];
    const float* xyz     = (const float*)d_in[1];
    const int*   nbr     = (const int*)d_in[2];
    const int*   mol_idx = (const int*)d_in[3];
    const float* embed   = (const float*)d_in[5];
    const float* We1     = (const float*)d_in[6];
    const float* be1     = (const float*)d_in[7];
    const float* We2     = (const float*)d_in[8];
    const float* be2     = (const float*)d_in[9];
    const float* Wn      = (const float*)d_in[10];
    const float* bn      = (const float*)d_in[11];
    const float* Wu1     = (const float*)d_in[12];
    const float* bu1     = (const float*)d_in[13];
    const float* Wu2     = (const float*)d_in[14];
    const float* bu2     = (const float*)d_in[15];
    const float* Wr1     = (const float*)d_in[16];
    const float* br1     = (const float*)d_in[17];
    const float* Wr2     = (const float*)d_in[18];
    const float* br2     = (const float*)d_in[19];
    float*       out     = (float*)d_out;

    // Workspace: r | rn(/tmp/h) | m   (3 x 30000*128 f32 = 46 MB)
    float* r  = (float*)d_ws;
    float* rn = r + (size_t)NATOMS * DD;
    float* m  = rn + (size_t)NATOMS * DD;

    const int rowtiles  = NATOMS / 16;   // 1875
    const int edgetiles = NEDGES / 16;   // 37500

    hipMemsetAsync(out, 0, (size_t)out_size * sizeof(float), stream);
    embed_kernel<<<(NATOMS * DD + 255) / 256, 256, 0, stream>>>(z, embed, r,
                                                                NATOMS * DD);

    for (int L = 0; L < 3; ++L) {
        // rn = r @ Wn[L] + bn[L]
        wmma_gemm_kernel<8, 4, 0, false><<<240, 128, 0, stream>>>(
            r, Wn + (size_t)L * DD * DD, bn + (size_t)L * DD, nullptr, rn,
            rowtiles);
        hipMemsetAsync(m, 0, (size_t)NATOMS * DD * sizeof(float), stream);
        // per-edge filter + bidirectional message scatter
        edge_kernel<<<1024, 128, 0, stream>>>(
            xyz, nbr, We1 + (size_t)L * GG * GG, be1 + (size_t)L * GG,
            We2 + (size_t)L * GG * DD, be2 + (size_t)L * DD, rn, m, edgetiles);
        // tmp = ssp(m @ Wu1[L] + bu1[L])   (reuse rn buffer)
        wmma_gemm_kernel<8, 4, 1, false><<<240, 128, 0, stream>>>(
            m, Wu1 + (size_t)L * DD * DD, bu1 + (size_t)L * DD, nullptr, rn,
            rowtiles);
        // r = r + tmp @ Wu2[L] + bu2[L]
        wmma_gemm_kernel<8, 4, 0, true><<<240, 128, 0, stream>>>(
            rn, Wu2 + (size_t)L * DD * DD, bu2 + (size_t)L * DD, r, r,
            rowtiles);
    }

    // h = ssp(r @ Wr1 + br1)  (30000 x 64, reuse rn buffer)
    wmma_gemm_kernel<4, 4, 1, false><<<240, 128, 0, stream>>>(r, Wr1, br1,
                                                              nullptr, rn,
                                                              rowtiles);
    // energy[mol] += h . Wr2 + br2
    readout_kernel<<<(NATOMS * 32 + 255) / 256, 256, 0, stream>>>(
        rn, Wr2, br2, mol_idx, out, NATOMS);
}
